// TreeTransformer_75170517615316
// MI455X (gfx1250) — matrix operand order repaired and checked by reading, not compiled, not run on hardware
//
#include <hip/hip_runtime.h>
#include <hip/hip_bf16.h>
#include <stdint.h>

typedef __bf16 bf16_t;
typedef __bf16 v16bf __attribute__((ext_vector_type(16)));
typedef float  v8f   __attribute__((ext_vector_type(8)));

namespace {
constexpr int SEQ    = 1024;
constexpr int BATCH  = 8;
constexpr int IDIM   = 256;
constexpr int HDIM   = 512;
constexpr int LAYERS = 4;
constexpr int NHEADS = 8;
constexpr int DHEAD  = 64;
constexpr int FFDIM  = 1024;
constexpr int MROWS  = BATCH * SEQ;   // 8192
constexpr float NEGB = -1.0e9f;
constexpr float EPSV = 1e-5f;
}

union Frag { v16bf v; uint4 u[2]; };

static __device__ __forceinline__ bf16_t f2bf(float f) {
  union { float f; uint32_t u; } c;
  c.f = f;
  uint32_t u = c.u;
  uint32_t r = (u + 0x7FFFu + ((u >> 16) & 1u)) >> 16;
  unsigned short us = (unsigned short)r;
  bf16_t h;
  __builtin_memcpy(&h, &us, sizeof(us));
  return h;
}

// Per-lane async copy: 16 bytes global -> LDS, bypassing VGPRs (ASYNCcnt path).
// Low 32 bits of a generic pointer to LDS are the wave-relative LDS byte offset
// (ISA 10.2: generic LDS address = {SHARED_BASE, addr[31:0]}).
static __device__ __forceinline__ void async_copy_b128(bf16_t* lds, const bf16_t* g) {
  uint32_t l = (uint32_t)(uintptr_t)lds;
  asm volatile("global_load_async_to_lds_b128 %0, %1, off"
               :: "v"(l), "v"(g) : "memory");
}
static __device__ __forceinline__ void wait_async0() {
  asm volatile("s_wait_asynccnt 0x0" ::: "memory");
}

// ---------------- elementwise f32 -> bf16 ----------------
__global__ __launch_bounds__(256)
void cvt_f32_bf16(const float* __restrict__ in, bf16_t* __restrict__ out, int n) {
  int i = blockIdx.x * blockDim.x + threadIdx.x;
  if (i < n) out[i] = f2bf(in[i]);
}

// nodes (S,B,I) f32  ->  xin (B*S, I) bf16   (the transpose(1,0,2))
__global__ __launch_bounds__(256)
void prep_input_kernel(const float* __restrict__ nodes, bf16_t* __restrict__ out) {
  int idx = blockIdx.x * blockDim.x + threadIdx.x;
  if (idx >= SEQ * BATCH * IDIM) return;
  int i = idx % IDIM;
  int t = idx / IDIM;
  int b = t % BATCH;
  int s = t / BATCH;
  out[((long)(b * SEQ + s)) * IDIM + i] = f2bf(nodes[idx]);
}

// ---------------- tiled bf16 WMMA GEMM:  C = act(A @ W^T + bias) ----------------
// A: (M,K) bf16 row-major.  W: (N,K) bf16 row-major (so both are K-contiguous).
// 128x128 block tile, 8 waves, each wave 32x64 (2x4 WMMA tiles), K-step 32.
// Tiles staged in LDS via global_load_async_to_lds_b128 (no staging VGPRs).
__global__ __launch_bounds__(256)
void gemm_bf16(const bf16_t* __restrict__ A, const bf16_t* __restrict__ W,
               const float* __restrict__ bias, float* __restrict__ outF,
               bf16_t* __restrict__ outB, int Mdim, int Ndim, int Kdim, int relu) {
  constexpr int LDT = 40;                     // padded bf16 stride (80B, 16B-aligned)
  __shared__ __attribute__((aligned(16))) bf16_t As[128 * LDT];
  __shared__ __attribute__((aligned(16))) bf16_t Ws[128 * LDT];
  const int m0 = blockIdx.y * 128;
  const int n0 = blockIdx.x * 128;
  const int t = threadIdx.x;
  const int lane = t & 31;
  const int wave = t >> 5;
  const int wr = (wave >> 1) << 5;            // 0,32,64,96
  const int wc = (wave & 1) << 6;             // 0,64
  const int lrow = lane & 15;
  const int alo = (lane < 16) ? 0 : 8;        // A-matrix K lo/hi half per lane group
  const int blo = (lane < 16) ? 0 : 8;        // B-matrix K lo/hi half per lane group
  const int srow  = t >> 1;                   // staging row 0..127
  const int shalf = (t & 1) << 4;             // staging col 0 or 16

  v8f acc[2][4] = {};

  for (int k0 = 0; k0 < Kdim; k0 += 32) {
    const bf16_t* ga = A + (long)(m0 + srow) * Kdim + k0 + shalf;
    const bf16_t* gw = W + (long)(n0 + srow) * Kdim + k0 + shalf;
    bf16_t* la = &As[srow * LDT + shalf];
    bf16_t* lw = &Ws[srow * LDT + shalf];
    async_copy_b128(la,     ga);
    async_copy_b128(la + 8, ga + 8);
    async_copy_b128(lw,     gw);
    async_copy_b128(lw + 8, gw + 8);
    if (k0 + 32 < Kdim) {
      __builtin_prefetch(ga + 32, 0, 3);
      __builtin_prefetch(gw + 32, 0, 3);
    }
    wait_async0();
    __syncthreads();

    Frag afr[2];
#pragma unroll
    for (int rt = 0; rt < 2; rt++) {
      const bf16_t* s = &As[(wr + rt * 16 + lrow) * LDT + alo];
      afr[rt].u[0] = *(const uint4*)s;          // K = lo..lo+7
      afr[rt].u[1] = *(const uint4*)(s + 16);   // K = lo+16..lo+23
    }
#pragma unroll
    for (int ct = 0; ct < 4; ct++) {
      Frag bfr;
      const bf16_t* s = &Ws[(wc + ct * 16 + lrow) * LDT + blo];
      bfr.u[0] = *(const uint4*)s;              // K = blo..blo+7
      bfr.u[1] = *(const uint4*)(s + 16);       // K = blo+16..blo+23
#pragma unroll
      for (int rt = 0; rt < 2; rt++)
        acc[rt][ct] = __builtin_amdgcn_wmma_f32_16x16x32_bf16(
            false, afr[rt].v, false, bfr.v, (short)0, acc[rt][ct], false, false);
    }
    __syncthreads();
  }

#pragma unroll
  for (int rt = 0; rt < 2; rt++)
#pragma unroll
    for (int ct = 0; ct < 4; ct++)
#pragma unroll
      for (int rr = 0; rr < 8; rr++) {
        int row = m0 + wr + rt * 16 + rr + ((lane >> 4) << 3);
        int col = n0 + wc + ct * 16 + (lane & 15);
        float v = acc[rt][ct][rr] + bias[col];
        if (relu) v = fmaxf(v, 0.0f);
        long o = (long)row * Ndim + col;
        if (outF) outF[o] = v;
        if (outB) outB[o] = f2bf(v);
      }
}

// qkv (M,3H) bf16 -> q,k (B,NH,S,HD) and vT (B,NH,HD,S)
__global__ __launch_bounds__(256)
void reorder_qkv_kernel(const bf16_t* __restrict__ qkv, bf16_t* __restrict__ q,
                        bf16_t* __restrict__ k, bf16_t* __restrict__ vt) {
  long idx = (long)blockIdx.x * blockDim.x + threadIdx.x;
  if (idx >= (long)MROWS * 3 * HDIM) return;
  int c = (int)(idx % (3 * HDIM));
  int m = (int)(idx / (3 * HDIM));
  int b = m >> 10, s = m & 1023;
  int which = c / HDIM;
  int hc = c % HDIM;
  int hh = hc >> 6, d = hc & 63;
  bf16_t val = qkv[idx];
  long bh = (long)(b * NHEADS + hh);
  if (which == 0)      q [(bh * SEQ + s) * DHEAD + d] = val;
  else if (which == 1) k [(bh * SEQ + s) * DHEAD + d] = val;
  else                 vt[(bh * DHEAD + d) * SEQ + s] = val;
}

// ---------------- fused flash attention (tree mask from dist) ----------------
// grid: (SEQ/128, B*NH), 256 threads = 8 waves, each wave owns one 16-query tile.
__global__ __launch_bounds__(256)
void attn_kernel(const bf16_t* __restrict__ q, const bf16_t* __restrict__ kk,
                 const bf16_t* __restrict__ vt, const int* __restrict__ dist,
                 bf16_t* __restrict__ out) {
  __shared__ __attribute__((aligned(16))) bf16_t pbuf[8 * 16 * 32];
  const int bh = blockIdx.y;
  const int b = bh >> 3, h = bh & 7;
  const int wave = threadIdx.x >> 5, lane = threadIdx.x & 31;
  const int q0 = blockIdx.x * 128 + wave * 16;
  const int lrow = lane & 15;
  const int lqoff = (lane >> 4) << 3;        // C/D row offset: 0 (lanes<16) or 8
  const int alo = (lane < 16) ? 0 : 8;
  const int blo = (lane < 16) ? 0 : 8;
  const bf16_t* Qp = q  + ((long)bh * SEQ + q0) * DHEAD;
  const bf16_t* Kp = kk + (long)bh * SEQ * DHEAD;
  const bf16_t* Vt = vt + (long)bh * DHEAD * SEQ;
  const int* Drow0 = dist + (long)b * SEQ * SEQ;    // dist[b][0][*] for global heads

  Frag aq[2];
#pragma unroll
  for (int c = 0; c < 2; c++) {
    const bf16_t* s = Qp + lrow * DHEAD + c * 32 + alo;
    aq[c].u[0] = *(const uint4*)s;
    aq[c].u[1] = *(const uint4*)(s + 16);
  }
  float m8[8], l8[8];
  v8f oacc[4] = {};
#pragma unroll
  for (int r = 0; r < 8; r++) { m8[r] = -1.0e30f; l8[r] = 0.0f; }
  bf16_t* pw = pbuf + wave * (16 * 32);

  for (int kb = 0; kb < SEQ; kb += 32) {
    // ---- scores: Q(16x64) @ K^T(64x32) as 2x2 WMMA ----
    v8f sc[2] = {};
#pragma unroll
    for (int ks = 0; ks < 2; ks++)
#pragma unroll
      for (int c = 0; c < 2; c++) {
        Frag bk;
        const bf16_t* s = Kp + (long)(kb + ks * 16 + lrow) * DHEAD + c * 32 + blo;
        bk.u[0] = *(const uint4*)s;
        bk.u[1] = *(const uint4*)(s + 16);
        sc[ks] = __builtin_amdgcn_wmma_f32_16x16x32_bf16(
            false, aq[c].v, false, bk.v, (short)0, sc[ks], false, false);
      }
    // ---- tree mask bias ----
#pragma unroll
    for (int ks = 0; ks < 2; ks++) {
      int ki = kb + ks * 16 + lrow;
      int g0 = Drow0[ki];
      const int* Dcol = dist + (long)b * SEQ * SEQ + ki;
#pragma unroll
      for (int rr = 0; rr < 8; rr++) {
        int qi = q0 + rr + lqoff;
        int dv = Dcol[(long)qi * SEQ];
        bool allowed = (h < 4) ? (dv == 1) : (h < 6) ? (dv >= 1) : (g0 >= 1);
        allowed = allowed || (qi == ki);
        sc[ks][rr] = sc[ks][rr] * 0.125f + (allowed ? 0.0f : NEGB);
      }
    }
    // ---- online softmax (row = C/D VGPR index; columns live on 16-lane halves) ----
#pragma unroll
    for (int rr = 0; rr < 8; rr++) {
      float mx = fmaxf(sc[0][rr], sc[1][rr]);
#pragma unroll
      for (int off = 8; off >= 1; off >>= 1)
        mx = fmaxf(mx, __shfl_xor(mx, off, 32));
      float mn = fmaxf(m8[rr], mx);
      float corr = __expf(m8[rr] - mn);
      m8[rr] = mn;
      float p0 = __expf(sc[0][rr] - mn);
      float p1 = __expf(sc[1][rr] - mn);
      sc[0][rr] = p0; sc[1][rr] = p1;
      float rs = p0 + p1;
#pragma unroll
      for (int off = 8; off >= 1; off >>= 1)
        rs += __shfl_xor(rs, off, 32);
      l8[rr] = l8[rr] * corr + rs;
#pragma unroll
      for (int ct = 0; ct < 4; ct++) oacc[ct][rr] *= corr;
    }
    // ---- P: C/D layout -> A layout via LDS ----
#pragma unroll
    for (int ks = 0; ks < 2; ks++)
#pragma unroll
      for (int rr = 0; rr < 8; rr++)
        pw[(rr + lqoff) * 32 + ks * 16 + lrow] = f2bf(sc[ks][rr]);
    asm volatile("s_wait_dscnt 0x0" ::: "memory");
    Frag ap;
    {
      const bf16_t* s = pw + lrow * 32 + alo;
      ap.u[0] = *(const uint4*)s;
      ap.u[1] = *(const uint4*)(s + 16);
    }
    // ---- O += P(16x32) @ V(32x64) as 4 WMMA (V^T rows are S-contiguous) ----
#pragma unroll
    for (int ct = 0; ct < 4; ct++) {
      Frag bv;
      const bf16_t* s = Vt + (long)(ct * 16 + lrow) * SEQ + kb + blo;
      bv.u[0] = *(const uint4*)s;
      bv.u[1] = *(const uint4*)(s + 16);
      oacc[ct] = __builtin_amdgcn_wmma_f32_16x16x32_bf16(
          false, ap.v, false, bv.v, (short)0, oacc[ct], false, false);
    }
  }
  // ---- normalize and scatter back to (B*S, H) bf16 ----
#pragma unroll
  for (int ct = 0; ct < 4; ct++)
#pragma unroll
    for (int rr = 0; rr < 8; rr++) {
      int qi = q0 + rr + lqoff;
      int d = ct * 16 + lrow;
      float ov = oacc[ct][rr] / l8[rr];
      out[((long)(b * SEQ + qi)) * HDIM + h * DHEAD + d] = f2bf(ov);
    }
}

// ---------------- LayerNorm (optionally with residual add); wave per row ----------------
__global__ __launch_bounds__(256)
void ln_kernel(const float* __restrict__ x, const float* __restrict__ res,
               const float* __restrict__ g, const float* __restrict__ bt,
               float* __restrict__ outF, bf16_t* __restrict__ outB, int Mrows) {
  int row = blockIdx.x * 8 + (threadIdx.x >> 5);
  int lane = threadIdx.x & 31;
  if (row >= Mrows) return;
  const float* xr = x + (long)row * HDIM;
  const float* rr = res ? res + (long)row * HDIM : nullptr;
  float v[16];
  float s = 0.0f;
#pragma unroll
  for (int i = 0; i < 16; i++) {
    int col = lane + i * 32;
    float t = xr[col];
    if (rr) t += rr[col];
    v[i] = t; s += t;
  }
#pragma unroll
  for (int off = 16; off >= 1; off >>= 1) s += __shfl_xor(s, off, 32);
  float mean = s * (1.0f / HDIM);
  float var = 0.0f;
#pragma unroll
  for (int i = 0; i < 16; i++) { float d = v[i] - mean; var += d * d; }
#pragma unroll
  for (int off = 16; off >= 1; off >>= 1) var += __shfl_xor(var, off, 32);
  var *= (1.0f / HDIM);
  float inv = rsqrtf(var + EPSV);
#pragma unroll
  for (int i = 0; i < 16; i++) {
    int col = lane + i * 32;
    float y = (v[i] - mean) * inv * g[col] + bt[col];
    outF[(long)row * HDIM + col] = y;
    outB[(long)row * HDIM + col] = f2bf(y);
  }
}

// ---------------- final batchnorm over B of x[:,0,:] ----------------
__global__ __launch_bounds__(256)
void bn_kernel(const float* __restrict__ x, const float* __restrict__ g,
               const float* __restrict__ bt, float* __restrict__ out) {
  int hcol = blockIdx.x * blockDim.x + threadIdx.x;
  if (hcol >= HDIM) return;
  float s = 0.0f, s2 = 0.0f;
#pragma unroll
  for (int bb = 0; bb < BATCH; bb++) {
    float t = x[((long)bb * SEQ) * HDIM + hcol];
    s += t; s2 += t * t;
  }
  float mean = s * (1.0f / BATCH);
  float var = s2 * (1.0f / BATCH) - mean * mean;
  float inv = rsqrtf(var + EPSV);
#pragma unroll
  for (int bb = 0; bb < BATCH; bb++) {
    float t = x[((long)bb * SEQ) * HDIM + hcol];
    out[bb * HDIM + hcol] = (t - mean) * inv * g[hcol] + bt[hcol];
  }
}

extern "C" void kernel_launch(void* const* d_in, const int* in_sizes, int n_in,
                              void* d_out, int out_size, void* d_ws, size_t ws_size,
                              hipStream_t stream) {
  (void)in_sizes; (void)n_in; (void)out_size; (void)ws_size;
  const float* nodes   = (const float*)d_in[0];
  const int*   dist    = (const int*)d_in[1];
  const float* dense_w = (const float*)d_in[2];
  const float* dense_b = (const float*)d_in[3];
  const float* dln_g   = (const float*)d_in[4];
  const float* dln_b   = (const float*)d_in[5];
  const float* qkv_w   = (const float*)d_in[6];
  const float* qkv_b   = (const float*)d_in[7];
  const float* out_w   = (const float*)d_in[8];
  const float* out_b   = (const float*)d_in[9];
  const float* ln1_g   = (const float*)d_in[10];
  const float* ln1_b   = (const float*)d_in[11];
  const float* f1_w    = (const float*)d_in[12];
  const float* f1_b    = (const float*)d_in[13];
  const float* f2_w    = (const float*)d_in[14];
  const float* f2_b    = (const float*)d_in[15];
  const float* ln2_g   = (const float*)d_in[16];
  const float* ln2_b   = (const float*)d_in[17];
  const float* bn_g    = (const float*)d_in[18];
  const float* bn_b    = (const float*)d_in[19];

  char* ws = (char*)d_ws;
  size_t off = 0;
  auto bump = [&](size_t bytes) -> void* {
    void* p = ws + off;
    off += (bytes + 255) & ~(size_t)255;
    return p;
  };
  bf16_t* xin_bf  = (bf16_t*)bump((size_t)MROWS * IDIM * 2);
  bf16_t* dw_bf   = (bf16_t*)bump((size_t)HDIM * IDIM * 2);
  bf16_t* qkvw_bf = (bf16_t*)bump((size_t)LAYERS * 3 * HDIM * HDIM * 2);
  bf16_t* outw_bf = (bf16_t*)bump((size_t)LAYERS * HDIM * HDIM * 2);
  bf16_t* f1w_bf  = (bf16_t*)bump((size_t)LAYERS * FFDIM * HDIM * 2);
  bf16_t* f2w_bf  = (bf16_t*)bump((size_t)LAYERS * HDIM * FFDIM * 2);
  float*  x_f     = (float*)bump((size_t)MROWS * HDIM * 4);
  float*  o_f     = (float*)bump((size_t)MROWS * HDIM * 4);
  bf16_t* x_bf    = (bf16_t*)bump((size_t)MROWS * HDIM * 2);
  bf16_t* qkv_bf  = (bf16_t*)bump((size_t)MROWS * 3 * HDIM * 2);
  bf16_t* q_bf    = (bf16_t*)bump((size_t)MROWS * HDIM * 2);
  bf16_t* k_bf    = (bf16_t*)bump((size_t)MROWS * HDIM * 2);
  bf16_t* vt_bf   = (bf16_t*)bump((size_t)MROWS * HDIM * 2);
  bf16_t* at_bf   = (bf16_t*)bump((size_t)MROWS * HDIM * 2);
  bf16_t* mid_bf  = (bf16_t*)bump((size_t)MROWS * FFDIM * 2);

  auto cvt = [&](const float* src, bf16_t* dst, int n) {
    cvt_f32_bf16<<<(n + 255) / 256, 256, 0, stream>>>(src, dst, n);
  };
  cvt(dense_w, dw_bf, HDIM * IDIM);
  cvt(qkv_w, qkvw_bf, LAYERS * 3 * HDIM * HDIM);
  cvt(out_w, outw_bf, LAYERS * HDIM * HDIM);
  cvt(f1_w, f1w_bf, LAYERS * FFDIM * HDIM);
  cvt(f2_w, f2w_bf, LAYERS * HDIM * FFDIM);
  prep_input_kernel<<<(SEQ * BATCH * IDIM + 255) / 256, 256, 0, stream>>>(nodes, xin_bf);

  auto gemm = [&](const bf16_t* A, const bf16_t* W, const float* bias,
                  float* oF, bf16_t* oB, int N2, int K2, int relu) {
    dim3 grid(N2 / 128, MROWS / 128);
    gemm_bf16<<<grid, 256, 0, stream>>>(A, W, bias, oF, oB, MROWS, N2, K2, relu);
  };

  // dense in-projection + LN
  gemm(xin_bf, dw_bf, dense_b, o_f, nullptr, HDIM, IDIM, 0);
  ln_kernel<<<MROWS / 8, 256, 0, stream>>>(o_f, nullptr, dln_g, dln_b, x_f, x_bf, MROWS);

  for (int l = 0; l < LAYERS; l++) {
    gemm(x_bf, qkvw_bf + (size_t)l * 3 * HDIM * HDIM, qkv_b + (size_t)l * 3 * HDIM,
         nullptr, qkv_bf, 3 * HDIM, HDIM, 0);
    reorder_qkv_kernel<<<(MROWS * 3 * HDIM + 255) / 256, 256, 0, stream>>>(
        qkv_bf, q_bf, k_bf, vt_bf);
    attn_kernel<<<dim3(SEQ / 128, BATCH * NHEADS), 256, 0, stream>>>(
        q_bf, k_bf, vt_bf, dist, at_bf);
    gemm(at_bf, outw_bf + (size_t)l * HDIM * HDIM, out_b + (size_t)l * HDIM,
         o_f, nullptr, HDIM, HDIM, 0);
    ln_kernel<<<MROWS / 8, 256, 0, stream>>>(x_f, o_f, ln1_g + (size_t)l * HDIM,
                                             ln1_b + (size_t)l * HDIM, x_f, x_bf, MROWS);
    gemm(x_bf, f1w_bf + (size_t)l * FFDIM * HDIM, f1_b + (size_t)l * FFDIM,
         nullptr, mid_bf, FFDIM, HDIM, 1);
    gemm(mid_bf, f2w_bf + (size_t)l * HDIM * FFDIM, f2_b + (size_t)l * HDIM,
         o_f, nullptr, HDIM, FFDIM, 0);
    ln_kernel<<<MROWS / 8, 256, 0, stream>>>(x_f, o_f, ln2_g + (size_t)l * HDIM,
                                             ln2_b + (size_t)l * HDIM, x_f, x_bf, MROWS);
  }
  bn_kernel<<<(HDIM + 255) / 256, 256, 0, stream>>>(x_f, bn_g, bn_b, (float*)d_out);
}
